// RandomAttention_88648124991336
// MI455X (gfx1250) — matrix operand order repaired and checked by reading, not compiled, not run on hardware
//
#include <hip/hip_runtime.h>
#include <hip/hip_bf16.h>

typedef __attribute__((ext_vector_type(16))) __bf16 v16bf;
typedef __attribute__((ext_vector_type(8)))  float  v8f;

union AFrag { v16bf v; __bf16 h[16]; unsigned int u[8]; };
union F8    { v8f  v; float  f[8]; };

#define WMMA_BF16(A, B, C) \
  __builtin_amdgcn_wmma_f32_16x16x32_bf16(false, (A), false, (B), (short)0, (C), false, false)

__device__ inline float hmax16(float v) {
#pragma unroll
  for (int m = 1; m < 16; m <<= 1) v = fmaxf(v, __shfl_xor(v, m, 32));
  return v;
}
__device__ inline float hsum16(float v) {
#pragma unroll
  for (int m = 1; m < 16; m <<= 1) v += __shfl_xor(v, m, 32);
  return v;
}

// ---------------------------------------------------------------------------
// sel[b][n]: 1.0 for CLS (n==0) and for selected patch tokens, else 0.0
// ---------------------------------------------------------------------------
__global__ void sel_init_kernel(float* sel) {
  int i = blockIdx.x * blockDim.x + threadIdx.x;
  if (i < 4 * 1024) sel[i] = ((i & 1023) == 0) ? 1.0f : 0.0f;
}

__global__ void sel_scatter_kernel(float* sel, const int* __restrict__ idx, int k,
                                   const int* __restrict__ counter,
                                   const int* __restrict__ renable) {
  if (!(renable[0] != 0 && counter[0] > 50)) return;
  int i = blockIdx.x * blockDim.x + threadIdx.x;
  if (i < 4 * k) {
    int b = i / k;
    int tok = idx[i] + 1;  // shift past CLS
    sel[b * 1024 + tok] = 1.0f;
  }
}

// ---------------------------------------------------------------------------
// Y[M,Nn] = X[M,K] @ W[K,Nn] (+bias), f32 in/out, BF16 WMMA, f32 accumulate.
// Block: 256 thr = 8 waves; block tile 128x128, BK=32; wave tile 32x64
// (2 A-frags x 4 B-frags -> 8 WMMAs per K-step per wave).
// ---------------------------------------------------------------------------
#define GBM 128
#define GBN 128
#define GBK 32

__global__ __launch_bounds__(256) void gemm_bf16_kernel(
    const float* __restrict__ X, const float* __restrict__ W,
    const float* __restrict__ bias, float* __restrict__ Y,
    int M, int K, int Nn) {
  __shared__ __bf16 As[GBM][GBK + 2];   // [m][k]
  __shared__ __bf16 Bs[GBN][GBK + 2];   // [n][k] (transposed for B-frags)

  const int tid  = threadIdx.x;
  const int lane = tid & 31;
  const int wid  = tid >> 5;     // 0..7
  const int wy   = wid & 3;      // M sub-tile: 32*wy
  const int wx   = wid >> 2;     // N sub-tile: 64*wx
  const int g    = lane >> 4;
  const int ln   = lane & 15;
  const int mBase = blockIdx.y * GBM;
  const int nBase = blockIdx.x * GBN;

  // staging maps (fully coalesced float4 global loads)
  const int xr = tid >> 1;           // 0..127 : X row
  const int xc = (tid & 1) * 16;     // 0,16   : X k segment
  const int wk = tid >> 3;           // 0..31  : W k row
  const int wn = (tid & 7) * 16;     // 0..112 : W n segment

  F8 acc[2][4];
#pragma unroll
  for (int s = 0; s < 2; s++)
#pragma unroll
    for (int t = 0; t < 4; t++)
#pragma unroll
      for (int r = 0; r < 8; r++) acc[s][t].f[r] = 0.0f;

  for (int k0 = 0; k0 < K; k0 += GBK) {
    const float* xp = X + (size_t)(mBase + xr) * K + k0 + xc;
    const float* wp = W + (size_t)(k0 + wk) * Nn + nBase + wn;
#pragma unroll
    for (int i = 0; i < 4; i++) {
      float4 f = ((const float4*)xp)[i];
      As[xr][xc + 4 * i + 0] = (__bf16)f.x;
      As[xr][xc + 4 * i + 1] = (__bf16)f.y;
      As[xr][xc + 4 * i + 2] = (__bf16)f.z;
      As[xr][xc + 4 * i + 3] = (__bf16)f.w;
    }
#pragma unroll
    for (int i = 0; i < 4; i++) {
      float4 f = ((const float4*)wp)[i];
      Bs[wn + 4 * i + 0][wk] = (__bf16)f.x;
      Bs[wn + 4 * i + 1][wk] = (__bf16)f.y;
      Bs[wn + 4 * i + 2][wk] = (__bf16)f.z;
      Bs[wn + 4 * i + 3][wk] = (__bf16)f.w;
    }
    if (k0 + GBK < K) {  // gfx1250 global_prefetch_b8 for the next K tile
      __builtin_prefetch(xp + GBK, 0, 1);
      __builtin_prefetch(wp + (size_t)GBK * Nn, 0, 1);
    }
    __syncthreads();

    AFrag a0, a1, bf;
    const int am = wy * 32 + ln;
#pragma unroll
    for (int v = 0; v < 8; v++) {
      int kk = (v & 3) * 2 + (v >> 2) * 16 + 8 * g;  // 16-bit A 16x32 layout
      a0.u[v] = *(const unsigned int*)&As[am][kk];
      a1.u[v] = *(const unsigned int*)&As[am + 16][kk];
    }
#pragma unroll
    for (int t = 0; t < 4; t++) {
      const int bn = wx * 64 + 16 * t + ln;
#pragma unroll
      for (int v = 0; v < 8; v++)
        bf.u[v] = *(const unsigned int*)&Bs[bn][(v & 3) * 2 + (v >> 2) * 16 + 8 * g];  // B 32x16 layout
      acc[0][t].v = WMMA_BF16(a0.v, bf.v, acc[0][t].v);
      acc[1][t].v = WMMA_BF16(a1.v, bf.v, acc[1][t].v);
    }
    __syncthreads();
  }

#pragma unroll
  for (int t = 0; t < 4; t++) {
    const int col = nBase + wx * 64 + 16 * t + ln;
    const float bv = bias ? bias[col] : 0.0f;
#pragma unroll
    for (int s = 0; s < 2; s++) {
#pragma unroll
      for (int r = 0; r < 8; r++) {
        int row = mBase + wy * 32 + s * 16 + r + 8 * g;
        Y[(size_t)row * Nn + col] = acc[s][t].f[r] + bv;
      }
    }
  }
}

// ---------------------------------------------------------------------------
// Fused masked flash attention. Grid: (B*H, N/64); block 128 thr = 4 waves,
// each wave owns 16 query rows. Streams K/V in 32-key blocks through LDS.
//   ctx[i] = sum_j e^{s_ij-m} * m_j * v_j / (sum_j e*m_j + eps*sum_j e)
//   m_j = sel_i ? 1 : sel_j  (enabled path); m_j = 1 (disabled path, no eps)
// ---------------------------------------------------------------------------
#define AN 1024
#define EPSV 1e-8f

__global__ __launch_bounds__(128) void attn_kernel(
    const float* __restrict__ qkv, const float* __restrict__ sel,
    float* __restrict__ ctx, const int* __restrict__ counter,
    const int* __restrict__ renable) {
  __shared__ __bf16 Ks[32][64 + 2];       // [key][d]
  __shared__ __bf16 Vt[64][32 + 2];       // [d][key] (transposed for B-frags)
  __shared__ __bf16 Ps[4][16][32 + 2];    // per-wave masked-P staging

  const int bh = blockIdx.x;
  const int b = bh >> 4, h = bh & 15;
  const int tid = threadIdx.x, lane = tid & 31, wid = tid >> 5;
  const int g = lane >> 4, ln = lane & 15;
  const int row0 = blockIdx.y * 64 + wid * 16;
  const bool enabled = (renable[0] != 0) && (counter[0] > 50);
  const float scale = 0.125f;  // Dh=64 -> 1/sqrt(64)

  // Q A-fragments (d 0..31 and 32..63), bf16
  AFrag aq0, aq1;
  {
    const float* qp = qkv + ((size_t)(b * AN + row0 + ln) * 3072 + h * 64);
#pragma unroll
    for (int v = 0; v < 8; v++) {
      int d = (v & 3) * 2 + (v >> 2) * 16 + 8 * g;
      float2 f0 = *(const float2*)(qp + d);
      float2 f1 = *(const float2*)(qp + 32 + d);
      aq0.h[2 * v] = (__bf16)f0.x; aq0.h[2 * v + 1] = (__bf16)f0.y;
      aq1.h[2 * v] = (__bf16)f1.x; aq1.h[2 * v + 1] = (__bf16)f1.y;
    }
  }

  float rsel[8], mrun[8], lfull[8], lmask[8];
  F8 cacc[4];
#pragma unroll
  for (int r = 0; r < 8; r++) {
    rsel[r] = sel[b * AN + row0 + r + 8 * g];
    mrun[r] = -1e30f; lfull[r] = 0.0f; lmask[r] = 0.0f;
#pragma unroll
    for (int t = 0; t < 4; t++) cacc[t].f[r] = 0.0f;
  }

  for (int jb = 0; jb < AN; jb += 32) {
    {  // cooperative K/V stage: 128 thr x 16 f32 each
      const int jloc = tid & 31;
      const int dbase = (tid >> 5) * 16;
      const float* kp = qkv + ((size_t)(b * AN + jb + jloc) * 3072 + 1024 + h * 64 + dbase);
      const float* vp = qkv + ((size_t)(b * AN + jb + jloc) * 3072 + 2048 + h * 64 + dbase);
#pragma unroll
      for (int i = 0; i < 16; i++) {
        Ks[jloc][dbase + i] = (__bf16)kp[i];
        Vt[dbase + i][jloc] = (__bf16)vp[i];
      }
      if (jb + 32 < AN) {
        __builtin_prefetch(kp + (size_t)32 * 3072, 0, 1);
        __builtin_prefetch(vp + (size_t)32 * 3072, 0, 1);
      }
    }
    __syncthreads();

    // S = Q @ K^T for two 16-key tiles, K-dim 64 = 2 bf16 WMMAs each
    F8 s0, s1;
#pragma unroll
    for (int r = 0; r < 8; r++) { s0.f[r] = 0.0f; s1.f[r] = 0.0f; }
    AFrag bk;
#pragma unroll
    for (int c = 0; c < 2; c++) {
#pragma unroll
      for (int v = 0; v < 8; v++)
        bk.u[v] = *(const unsigned int*)&Ks[ln][32 * c + (v & 3) * 2 + (v >> 2) * 16 + 8 * g];
      s0.v = WMMA_BF16(c ? aq1.v : aq0.v, bk.v, s0.v);
    }
#pragma unroll
    for (int c = 0; c < 2; c++) {
#pragma unroll
      for (int v = 0; v < 8; v++)
        bk.u[v] = *(const unsigned int*)&Ks[16 + ln][32 * c + (v & 3) * 2 + (v >> 2) * 16 + 8 * g];
      s1.v = WMMA_BF16(c ? aq1.v : aq0.v, bk.v, s1.v);
    }

    // online softmax + mask, per row r (row = row0 + r + 8g)
    const float cs0 = sel[b * AN + jb + ln];
    const float cs1 = sel[b * AN + jb + 16 + ln];
#pragma unroll
    for (int r = 0; r < 8; r++) {
      float a0 = s0.f[r] * scale, a1 = s1.f[r] * scale;
      float mt = hmax16(fmaxf(a0, a1));
      float mnew = fmaxf(mrun[r], mt);
      float resc = __expf(mrun[r] - mnew);
      mrun[r] = mnew;
      float e0 = __expf(a0 - mnew), e1 = __expf(a1 - mnew);
      lfull[r] = lfull[r] * resc + hsum16(e0 + e1);
      float w0, w1;
      if (enabled && rsel[r] == 0.0f) { w0 = e0 * cs0; w1 = e1 * cs1; }
      else                            { w0 = e0;       w1 = e1; }
      lmask[r] = lmask[r] * resc + hsum16(w0 + w1);
#pragma unroll
      for (int t = 0; t < 4; t++) cacc[t].f[r] *= resc;
      Ps[wid][r + 8 * g][ln]      = (__bf16)w0;
      Ps[wid][r + 8 * g][16 + ln] = (__bf16)w1;
    }
    asm volatile("s_wait_dscnt 0" ::: "memory");  // wave-local LDS RAW fence

    // ctx += P(16x32) @ V(32x64): 4 d-tiles
    AFrag ap, bv;
#pragma unroll
    for (int v = 0; v < 8; v++) {
      int kk = (v & 3) * 2 + (v >> 2) * 16 + 8 * g;
      ap.u[v] = *(const unsigned int*)&Ps[wid][ln][kk];
    }
#pragma unroll
    for (int t = 0; t < 4; t++) {
#pragma unroll
      for (int v = 0; v < 8; v++)
        bv.u[v] = *(const unsigned int*)&Vt[16 * t + ln][(v & 3) * 2 + (v >> 2) * 16 + 8 * g];
      cacc[t].v = WMMA_BF16(ap.v, bv.v, cacc[t].v);
    }
    __syncthreads();
  }

  // write ctx in [B,N,C] layout (head-merged) for the projection GEMM
#pragma unroll
  for (int r = 0; r < 8; r++) {
    float denom = enabled ? (lmask[r] + EPSV * lfull[r]) : lfull[r];
    float inv = 1.0f / denom;
    int row = row0 + r + 8 * g;
    float* op = ctx + ((size_t)(b * AN + row) * 1024 + h * 64);
#pragma unroll
    for (int t = 0; t < 4; t++) op[16 * t + ln] = cacc[t].f[r] * inv;
  }
}

// ---------------------------------------------------------------------------
extern "C" void kernel_launch(void* const* d_in, const int* in_sizes, int n_in,
                              void* d_out, int out_size, void* d_ws, size_t ws_size,
                              hipStream_t stream) {
  const float* x       = (const float*)d_in[0];
  const float* Wqkv    = (const float*)d_in[1];
  const float* Wproj   = (const float*)d_in[2];
  const float* bproj   = (const float*)d_in[3];
  const int*   pidx    = (const int*)d_in[4];
  const int*   counter = (const int*)d_in[5];
  const int*   renable = (const int*)d_in[6];
  float*       out     = (float*)d_out;

  // workspace: qkv [4096][3072] | ctx [4096][1024] | sel [4][1024]
  float* qkv = (float*)d_ws;
  float* ctx = qkv + (size_t)4096 * 3072;
  float* sel = ctx + (size_t)4096 * 1024;
  const int k = in_sizes[4] / 4;  // patches kept per batch (B=4)

  sel_init_kernel<<<16, 256, 0, stream>>>(sel);
  sel_scatter_kernel<<<(4 * k + 255) / 256, 256, 0, stream>>>(sel, pidx, k, counter, renable);

  dim3 gq(3072 / GBN, 4096 / GBM);
  gemm_bf16_kernel<<<gq, 256, 0, stream>>>(x, Wqkv, nullptr, qkv, 4096, 1024, 3072);

  dim3 ga(64, AN / 64);  // (B*H, row blocks)
  attn_kernel<<<ga, 128, 0, stream>>>(qkv, sel, ctx, counter, renable);

  dim3 gp(1024 / GBN, 4096 / GBM);
  gemm_bf16_kernel<<<gp, 256, 0, stream>>>(ctx, Wproj, bproj, out, 4096, 1024, 1024);
}
